// SelfAttentionME_36636071035186
// MI455X (gfx1250) — matrix operand order repaired and checked by reading, not compiled, not run on hardware
//
#include <hip/hip_runtime.h>
#include <hip/hip_bf16.h>

// SelfAttentionME for MI455X (gfx1250), flash-style two-phase attention.
// N=8192, C=256, CQK=32.  All intermediates f16 in workspace (L2-resident):
//   qh  [N][32]  f16   (row-major; serves both A- and B-fragments of q)
//   vT  [C][N]   f16   (transposed so PV B-fragments are contiguous)
//   m   [N]      f32   row max of energy
//   rZ  [N]      f32   1 / sum_j exp(energy[i,j] - m_i)
//
// k_attn: 4 waves/block share each 32x256 vT tile via DOUBLE-BUFFERED async
// global->LDS copies (ASYNCcnt path): tile it+1 is prefetched while the
// PV-WMMAs consume tile it.  One barrier per iteration.

typedef _Float16 half_t;
typedef __attribute__((ext_vector_type(16))) _Float16 v16h;
typedef __attribute__((ext_vector_type(8)))  _Float16 v8h;
typedef __attribute__((ext_vector_type(8)))  float    v8f;
typedef __attribute__((ext_vector_type(4)))  float    v4f;

#define N_PTS 8192
#define C_DIM 256
#define CQK_D 32
#define LDSW  48   // P-staging row stride (halfwords), multiple of 8 for 16B alignment
#define AW    4    // waves per block in k_attn
#define VBUF_BYTES (C_DIM * 32 * 2)   // 16KB per vT tile buffer

__device__ __forceinline__ v16h cat16(v8h lo, v8h hi) {
  return __builtin_shufflevector(lo, hi, 0,1,2,3,4,5,6,7,8,9,10,11,12,13,14,15);
}

__device__ __forceinline__ v8f wmma_f16(v16h a, v16h b, v8f c) {
  return __builtin_amdgcn_wmma_f32_16x16x32_f16(false, a, false, b, (short)0, c, false, false);
}

__device__ __forceinline__ v8f vzero8() {
  v8f z;
#pragma unroll
  for (int i = 0; i < 8; i++) z[i] = 0.0f;
  return z;
}

// A-fragment, f16 source. rowptr = &mat[row][0]; K-chunks {c0..c0+7, c0+16..c0+23},
// c0 = 0 (lanes 0-15) or 8 (lanes 16-31).  16B-aligned v8h loads.
__device__ __forceinline__ v16h fragA_h(const half_t* rowptr, int lane) {
  int c0 = (lane & 16) ? 8 : 0;
  v8h lo = *(const v8h*)(rowptr + c0);
  v8h hi = *(const v8h*)(rowptr + c0 + 16);
  return cat16(lo, hi);
}

// B-fragment, f16 source. rowptr = &mat[col][k0]; K-chunks {c0..c0+7, c0+16..c0+23},
// c0 = 0 (lanes 0-15) or 8 (lanes 16-31).
__device__ __forceinline__ v16h fragB_h(const half_t* rowptr, int lane) {
  const half_t* p = rowptr + ((lane & 16) ? 8 : 0);
  v8h lo = *(const v8h*)(p);
  v8h hi = *(const v8h*)(p + 16);
  return cat16(lo, hi);
}

// A-fragment from f32 memory (converted to f16 in-regs).
__device__ __forceinline__ v16h fragA_f(const float* rowptr, int lane) {
  int c0 = (lane & 16) ? 8 : 0;
  const v4f* p0 = (const v4f*)(rowptr + c0);
  const v4f* p1 = (const v4f*)(rowptr + c0 + 16);
  v4f a0 = p0[0], a1 = p0[1], b0 = p1[0], b1 = p1[1];
  v16h r;
#pragma unroll
  for (int i = 0; i < 4; i++) {
    r[i]      = (half_t)a0[i];
    r[4 + i]  = (half_t)a1[i];
    r[8 + i]  = (half_t)b0[i];
    r[12 + i] = (half_t)b1[i];
  }
  return r;
}

// B-fragment from f32 memory.
__device__ __forceinline__ v16h fragB_f(const float* rowptr, int lane) {
  const v4f* p = (const v4f*)(rowptr + ((lane & 16) ? 8 : 0));
  v4f a0 = p[0], a1 = p[1], a2 = p[4], a3 = p[5];
  v16h r;
#pragma unroll
  for (int i = 0; i < 4; i++) {
    r[i]      = (half_t)a0[i];
    r[4 + i]  = (half_t)a1[i];
    r[8 + i]  = (half_t)a2[i];
    r[12 + i] = (half_t)a3[i];
  }
  return r;
}

// ---------------------------------------------------------------------------
// Kernel 1: projections.  q = f @ Wqk^T  -> qh (f16 [N][32])
//                         v = f @ Wv^T+bv -> vT (f16 [C][N], transposed)
// ---------------------------------------------------------------------------
__global__ __launch_bounds__(64) void k_proj(
    const float* __restrict__ f, const float* __restrict__ Wqk,
    const float* __restrict__ Wv, const float* __restrict__ bv,
    half_t* __restrict__ qh, half_t* __restrict__ vT) {
  int lane  = threadIdx.x & 31;
  int wave  = threadIdx.x >> 5;
  int nbase = (blockIdx.x * 2 + wave) * 16;   // 512 tiles
  int mr    = lane & 15;
  int rofs  = (lane & 16) ? 8 : 0;

  v16h af[8];
#pragma unroll
  for (int kk = 0; kk < 8; kk++)
    af[kk] = fragA_f(f + (size_t)(nbase + mr) * C_DIM + kk * 32, lane);

#pragma unroll
  for (int ct = 0; ct < 2; ct++) {
    v8f acc = vzero8();
#pragma unroll
    for (int kk = 0; kk < 8; kk++) {
      v16h b = fragB_f(Wqk + (size_t)(ct * 16 + mr) * C_DIM + kk * 32, lane);
      acc = wmma_f16(af[kk], b, acc);
    }
#pragma unroll
    for (int r = 0; r < 8; r++)
      qh[(size_t)(nbase + rofs + r) * CQK_D + ct * 16 + mr] = (half_t)acc[r];
  }

#pragma unroll 1
  for (int ct = 0; ct < 16; ct++) {
    float bvc = bv[ct * 16 + mr];
    v8f acc;
#pragma unroll
    for (int r = 0; r < 8; r++) acc[r] = bvc;
#pragma unroll
    for (int kk = 0; kk < 8; kk++) {
      v16h b = fragB_f(Wv + (size_t)(ct * 16 + mr) * C_DIM + kk * 32, lane);
      acc = wmma_f16(af[kk], b, acc);
    }
    v8h pk;
#pragma unroll
    for (int r = 0; r < 8; r++) pk[r] = (half_t)acc[r];
    *(v8h*)(vT + (size_t)(ct * 16 + mr) * N_PTS + nbase + rofs) = pk;
  }
}

// ---------------------------------------------------------------------------
// Kernel 2: per-row softmax stats.  m_i = max_j q_i.q_j ; rZ_i = 1/sum exp(.-m_i)
// ---------------------------------------------------------------------------
__global__ __launch_bounds__(64) void k_stats(
    const half_t* __restrict__ qh, float* __restrict__ mrow, float* __restrict__ rzv) {
  int lane  = threadIdx.x & 31;
  int wave  = threadIdx.x >> 5;
  int ibase = (blockIdx.x * 2 + wave) * 16;
  int mr    = lane & 15;

  v16h a = fragA_h(qh + (size_t)(ibase + mr) * CQK_D, lane);

  float runM[8];
#pragma unroll
  for (int r = 0; r < 8; r++) runM[r] = -3.0e38f;

  for (int jt = 0; jt < N_PTS / 16; jt++) {
    v16h b = fragB_h(qh + (size_t)(jt * 16 + mr) * CQK_D, lane);
    v8f s = wmma_f16(a, b, vzero8());
#pragma unroll
    for (int r = 0; r < 8; r++) runM[r] = fmaxf(runM[r], s[r]);
  }
#pragma unroll
  for (int msk = 1; msk < 16; msk <<= 1) {
#pragma unroll
    for (int r = 0; r < 8; r++)
      runM[r] = fmaxf(runM[r], __shfl_xor(runM[r], msk, 32));
  }

  float runS[8];
#pragma unroll
  for (int r = 0; r < 8; r++) runS[r] = 0.0f;
  for (int jt = 0; jt < N_PTS / 16; jt++) {
    v16h b = fragB_h(qh + (size_t)(jt * 16 + mr) * CQK_D, lane);
    v8f s = wmma_f16(a, b, vzero8());
#pragma unroll
    for (int r = 0; r < 8; r++) runS[r] += __expf(s[r] - runM[r]);
  }
#pragma unroll
  for (int msk = 1; msk < 16; msk <<= 1) {
#pragma unroll
    for (int r = 0; r < 8; r++) runS[r] += __shfl_xor(runS[r], msk, 32);
  }

  if (mr == 0) {
    int rofs = (lane & 16) ? 8 : 0;
#pragma unroll
    for (int r = 0; r < 8; r++) {
      mrow[ibase + rofs + r] = runM[r];
      rzv[ibase + rofs + r]  = 1.0f / runS[r];
    }
  }
}

// ---------------------------------------------------------------------------
// Kernel 3: out[j,c] = gamma * sum_i exp(q_i.q_j - m_i)*rZ_i * v[i,c] + f[j,c]
// Block = AW waves, each owning one 16-row j-tile + 16x256 f32 accumulator.
// Double-buffered pipeline, one barrier per iteration:
//   [S-WMMA + exp + P^T staging for tile it]           (no vbuf use)
//   s_wait_asynccnt 0 ; __syncthreads()                (tile it landed)
//   issue async copy of tile it+1 -> other buffer      (full-iter prefetch)
//   16 PV-WMMAs reading B-fragments from buffer it&1
// ---------------------------------------------------------------------------
__global__ __launch_bounds__(32 * AW) void k_attn(
    const half_t* __restrict__ qh, const half_t* __restrict__ vT,
    const float* __restrict__ mrow, const float* __restrict__ rzv,
    const float* __restrict__ f, const float* __restrict__ gamma,
    float* __restrict__ out) {
  __shared__ __attribute__((aligned(16))) half_t vbuf[2][C_DIM * 32]; // 2x16KB
  __shared__ __attribute__((aligned(16))) half_t pst[AW][16 * LDSW];  // P^T staging

  int tid   = threadIdx.x;
  int lane  = tid & 31;
  int wave  = tid >> 5;
  int jbase = (blockIdx.x * AW + wave) * 16;
  int mr    = lane & 15;
  int rofs  = (lane & 16) ? 8 : 0;
  half_t* st = pst[wave];

  v16h aq = fragA_h(qh + (size_t)(jbase + mr) * CQK_D, lane);

  v8f acc[16];
#pragma unroll
  for (int ct = 0; ct < 16; ct++) acc[ct] = vzero8();

  // Async-copy plumbing: 1024 16B chunks per tile / (32*AW) threads.
  // chunk -> c = chunk>>2, sub = chunk&3; LDS halfword index = chunk*8.
  const int NCH = 1024 / (32 * AW);
  unsigned ldsoff[NCH];
  const half_t* gsrc[NCH];
#pragma unroll
  for (int t = 0; t < NCH; t++) {
    int chunk = tid + t * (32 * AW);
    ldsoff[t] = (unsigned)(size_t)(&vbuf[0][chunk * 8]);
    gsrc[t]   = vT + (size_t)(chunk >> 2) * N_PTS + (chunk & 3) * 8;
  }

  // Prologue: fetch tile 0 into buffer 0.
#pragma unroll
  for (int t = 0; t < NCH; t++) {
    asm volatile("global_load_async_to_lds_b128 %0, %1, off"
                 :: "v"(ldsoff[t]), "v"(gsrc[t]) : "memory");
    gsrc[t] += 32;   // advance one i-step (32 halfwords = 64B)
  }

  const int NIT = N_PTS / 32;
  for (int it = 0; it < NIT; it++) {
    int i0 = it * 32;

    // Build P^T tile (16 j x 32 i) in this wave's staging LDS (no vbuf use).
#pragma unroll
    for (int s = 0; s < 2; s++) {
      int ib = i0 + s * 16;
      v16h bq = fragB_h(qh + (size_t)(ib + mr) * CQK_D, lane);
      v8f S = wmma_f16(aq, bq, vzero8());     // S[M=j', N=i'] = q_j . q_i
      float mi = mrow[ib + mr];
      float ri = rzv[ib + mr];
#pragma unroll
      for (int r = 0; r < 8; r++) {
        float p = __expf(S[r] - mi) * ri;
        st[(rofs + r) * LDSW + s * 16 + mr] = (half_t)p;
      }
    }
    asm volatile("s_wait_dscnt 0" ::: "memory");   // cross-lane RAW in staging

    int c0 = (lane & 16) ? 8 : 0;
    v8h plo = *(const v8h*)(st + mr * LDSW + c0);
    v8h phi = *(const v8h*)(st + mr * LDSW + c0 + 16);
    v16h ap = cat16(plo, phi);

    // Tile it has landed in buffer it&1 (own loads done + block barrier).
    asm volatile("s_wait_asynccnt 0" ::: "memory");
    __syncthreads();

    // Prefetch tile it+1 into the other buffer (last read in iteration it-1,
    // so the barrier above already closed its WAR window).
    if (it + 1 < NIT) {
      unsigned bofs = ((it + 1) & 1) ? (unsigned)VBUF_BYTES : 0u;
#pragma unroll
      for (int t = 0; t < NCH; t++) {
        asm volatile("global_load_async_to_lds_b128 %0, %1, off"
                     :: "v"(ldsoff[t] + bofs), "v"(gsrc[t]) : "memory");
        gsrc[t] += 32;
      }
    }

    // PV-WMMAs from the current buffer.
    const half_t* vb = vbuf[it & 1];
#pragma unroll
    for (int ct = 0; ct < 16; ct++) {
      v16h bvf = fragB_h(vb + (ct * 16 + mr) * 32, lane);
      acc[ct] = wmma_f16(ap, bvf, acc[ct]);
    }
  }

  float g = gamma[0];
#pragma unroll
  for (int ct = 0; ct < 16; ct++) {
#pragma unroll
    for (int r = 0; r < 8; r++) {
      size_t idx = (size_t)(jbase + rofs + r) * C_DIM + ct * 16 + mr;
      out[idx] = g * acc[ct][r] + f[idx];
    }
  }
}

// ---------------------------------------------------------------------------
extern "C" void kernel_launch(void* const* d_in, const int* in_sizes, int n_in,
                              void* d_out, int out_size, void* d_ws, size_t ws_size,
                              hipStream_t stream) {
  const float* f     = (const float*)d_in[0];
  const float* Wqk   = (const float*)d_in[1];
  const float* Wv    = (const float*)d_in[2];
  const float* bv    = (const float*)d_in[3];
  const float* gamma = (const float*)d_in[4];
  float* out = (float*)d_out;

  char* ws = (char*)d_ws;
  size_t off = 0;
  half_t* qh = (half_t*)(ws + off); off += (size_t)N_PTS * CQK_D * sizeof(half_t);   // 512 KB
  half_t* vT = (half_t*)(ws + off); off += (size_t)C_DIM * N_PTS * sizeof(half_t);   // 4 MB
  float*  mr = (float*)(ws + off);  off += (size_t)N_PTS * sizeof(float);            // 32 KB
  float*  rz = (float*)(ws + off);  off += (size_t)N_PTS * sizeof(float);            // 32 KB

  k_proj <<<dim3(N_PTS / 32), dim3(64), 0, stream>>>(f, Wqk, Wv, bv, qh, vT);
  k_stats<<<dim3(N_PTS / 32), dim3(64), 0, stream>>>(qh, mr, rz);
  k_attn <<<dim3(N_PTS / (16 * AW)), dim3(32 * AW), 0, stream>>>(qh, vT, mr, rz, f, gamma, out);
}
